// Recurrent_50371376447522
// MI455X (gfx1250) — matrix operand order repaired and checked by reading, not compiled, not run on hardware
//
#include <hip/hip_runtime.h>
#include <math.h>

// Problem dims (fixed by the reference)
#define T_DIM   1024
#define B_DIM   64
#define IN_DIM  512
#define H_DIM   512
#define M_TOTAL (T_DIM * B_DIM)   // 65536 rows in the input-projection GEMM

typedef _Float16 v16h __attribute__((ext_vector_type(16)));
typedef float    v8f  __attribute__((ext_vector_type(8)));

// ---------------------------------------------------------------------------
// Kernel 1: xw = x @ Wxh^T + bxh  written straight into d_out [T*B, H] (fp32)
// One block (8 waves) per 16-row M-tile; wave w owns N columns [w*64, w*64+64).
// Each wave: 4 accumulators (16x16 tiles), A fragment reused across the 4.
// ---------------------------------------------------------------------------
__global__ __launch_bounds__(256) void xw_gemm_kernel(
    const float* __restrict__ x,    // [M_TOTAL, IN]
    const float* __restrict__ Wxh,  // [H, IN]
    const float* __restrict__ bxh,  // [H]
    float* __restrict__ out)        // [M_TOTAL, H]
{
    const int lane  = threadIdx.x & 31;
    const int wave  = threadIdx.x >> 5;         // 0..7 -> N group
    const int m0    = blockIdx.x * 16;
    const int nBase = wave * 64;
    const int half  = lane >> 4;                // 0 or 1
    const int l16   = lane & 15;

    v8f acc[4] = {};

    for (int kk = 0; kk < IN_DIM; kk += 32) {
        // A fragment (16x32 f16): row = m0+l16; elem i -> K = kk + half*16 + i
        v16h a;
        {
            const float* arow = x + (size_t)(m0 + l16) * IN_DIM + kk + half * 16;
            #pragma unroll
            for (int g = 0; g < 2; ++g)
                #pragma unroll
                for (int e = 0; e < 8; ++e)
                    a[g * 8 + e] = (_Float16)arow[g * 8 + e];
        }
        #pragma unroll
        for (int t = 0; t < 4; ++t) {
            // B fragment (32x16 f16): N = nBase+t*16+l16; elem i -> K = kk + half*16 + i
            const int n = nBase + t * 16 + l16;
            const float* brow = Wxh + (size_t)n * IN_DIM + kk + half * 16;
            v16h b;
            #pragma unroll
            for (int e = 0; e < 16; ++e)
                b[e] = (_Float16)brow[e];
            acc[t] = __builtin_amdgcn_wmma_f32_16x16x32_f16(
                false, a, false, b, (short)0, acc[t], false, false);
        }
    }

    // Epilogue: out[m][n] = acc + bxh[n]   (C/D layout: VGPR r -> M = half*8 + r)
    #pragma unroll
    for (int t = 0; t < 4; ++t) {
        const int n = nBase + t * 16 + l16;
        const float bias = bxh[n];
        #pragma unroll
        for (int r = 0; r < 8; ++r) {
            const int m = m0 + half * 8 + r;
            out[(size_t)m * H_DIM + n] = acc[t][r] + bias;
        }
    }
}

// ---------------------------------------------------------------------------
// Kernel 2: persistent recurrence.
//   h_t carried as f16 (WMMA input precision) double-buffered in d_ws;
//   Whh slices cached once in LDS as f16 (16 KB/wave, 130 KB/WG).
//   16 workgroups x 8 waves = 128 waves = 128 output tiles of h_new [64,512].
//   Device-wide monotonic atomic barrier between timesteps.
// ---------------------------------------------------------------------------
#define NWG          16
#define WAVES_PER_WG 8
#define LDS_PITCH    520   // 512 K-values + 8 pad (f16) per N-row -> bank spread

__device__ inline void grid_barrier(unsigned int* ctr, unsigned int target)
{
    __syncthreads();
    if (threadIdx.x == 0) {
        __threadfence();  // release this WG's h / out writes to device scope
        __hip_atomic_fetch_add(ctr, 1u, __ATOMIC_RELEASE, __HIP_MEMORY_SCOPE_AGENT);
        while (__hip_atomic_load(ctr, __ATOMIC_ACQUIRE, __HIP_MEMORY_SCOPE_AGENT) < target) {
            __builtin_amdgcn_s_sleep(1);
        }
    }
    __syncthreads();
}

__global__ __launch_bounds__(256) void rnn_persistent_kernel(
    const float* __restrict__ Whh,   // [H, H]
    const float* __restrict__ bhh,   // [H]
    float* __restrict__ out,         // [T, B, H]; holds xw+bxh on entry
    _Float16* __restrict__ hbuf0,    // [B, H] f16 scratch
    _Float16* __restrict__ hbuf1,    // [B, H] f16 scratch
    unsigned int* __restrict__ barrier_ctr)  // zeroed before launch
{
    __shared__ _Float16 ldsB[WAVES_PER_WG * 16 * LDS_PITCH];

    const int lane = threadIdx.x & 31;
    const int wave = threadIdx.x >> 5;
    const int tile = blockIdx.x * WAVES_PER_WG + wave;  // 0..127
    const int m0   = (tile >> 5) * 16;                  // batch-row tile
    const int n0   = (tile & 31) * 16;                  // hidden-col tile
    const int half = lane >> 4;
    const int l16  = lane & 15;

    // Stage this wave's Whh slice as f16 into LDS: myB[n][k], n in [0,16), k in [0,512)
    _Float16* myB = ldsB + wave * 16 * LDS_PITCH;
    for (int i = lane; i < 16 * H_DIM; i += 32) {
        const int n = i >> 9;          // 0..15
        const int k = i & (H_DIM - 1); // 0..511
        myB[n * LDS_PITCH + k] = (_Float16)Whh[(size_t)(n0 + n) * H_DIM + k];
    }
    __syncthreads();

    const float bias = bhh[n0 + l16];

    // ---- Step 0: h0 == 0, so h1 = tanh(xw_0 + bhh); no matmul needed. ----
    #pragma unroll
    for (int r = 0; r < 8; ++r) {
        const int m = m0 + half * 8 + r;
        const size_t oidx = (size_t)m * H_DIM + (n0 + l16);  // t = 0
        const float v = tanhf(out[oidx] + bias);
        out[oidx] = v;
        hbuf1[(size_t)m * H_DIM + (n0 + l16)] = (_Float16)v;  // h_1 in buf1
    }
    grid_barrier(barrier_ctr, (unsigned)NWG);

    // ---- Steps 1..T-1: out[t] = tanh(h_t @ Whh^T + bhh + xw_t) ----
    for (int t = 1; t < T_DIM; ++t) {
        const _Float16* hPrev = (t & 1) ? hbuf1 : hbuf0;  // holds h_t
        _Float16*       hCur  = (t & 1) ? hbuf0 : hbuf1;  // receives h_{t+1}

        v8f acc = {};
        for (int kk = 0; kk < H_DIM; kk += 32) {
            // A fragment from h_t: already f16, two contiguous 8-elem runs
            // elem i -> K = kk + half*16 + i
            v16h a;
            const _Float16* arow = hPrev + (size_t)(m0 + l16) * H_DIM + kk + half * 16;
            #pragma unroll
            for (int g = 0; g < 2; ++g)
                #pragma unroll
                for (int e = 0; e < 8; ++e)
                    a[g * 8 + e] = arow[g * 8 + e];

            // B fragment from LDS (f16): 16 contiguous elems per lane
            v16h b;
            const _Float16* brow = myB + l16 * LDS_PITCH + kk + half * 16;
            #pragma unroll
            for (int e = 0; e < 16; ++e)
                b[e] = brow[e];

            acc = __builtin_amdgcn_wmma_f32_16x16x32_f16(
                false, a, false, b, (short)0, acc, false, false);
        }

        const size_t tbase = (size_t)t * (B_DIM * H_DIM);
        #pragma unroll
        for (int r = 0; r < 8; ++r) {
            const int m = m0 + half * 8 + r;
            const size_t oidx = tbase + (size_t)m * H_DIM + (n0 + l16);
            const float v = tanhf(acc[r] + bias + out[oidx]);
            out[oidx] = v;
            hCur[(size_t)m * H_DIM + (n0 + l16)] = (_Float16)v;
        }
        grid_barrier(barrier_ctr, (unsigned)NWG * (unsigned)(t + 1));
    }
}

// ---------------------------------------------------------------------------
// Host entry
// ---------------------------------------------------------------------------
extern "C" void kernel_launch(void* const* d_in, const int* in_sizes, int n_in,
                              void* d_out, int out_size, void* d_ws, size_t ws_size,
                              hipStream_t stream)
{
    (void)in_sizes; (void)n_in; (void)out_size; (void)ws_size;

    const float* x   = (const float*)d_in[0];  // [T, B, IN]
    const float* Wxh = (const float*)d_in[1];  // [H, IN]
    const float* bxh = (const float*)d_in[2];  // [H]
    const float* Whh = (const float*)d_in[3];  // [H, H]
    const float* bhh = (const float*)d_in[4];  // [H]
    float* out = (float*)d_out;                // [T, B, H]

    _Float16* hbuf0 = (_Float16*)d_ws;
    _Float16* hbuf1 = hbuf0 + (size_t)B_DIM * H_DIM;   // +64 KB
    unsigned int* ctr =
        (unsigned int*)((char*)d_ws + 2 * (size_t)B_DIM * H_DIM * sizeof(_Float16));

    // Zero the barrier counter (capture-legal async memset).
    hipMemsetAsync(ctr, 0, sizeof(unsigned int), stream);

    // Phase 1: xw + bxh -> d_out (one block per 16-row M tile)
    xw_gemm_kernel<<<M_TOTAL / 16, 256, 0, stream>>>(x, Wxh, bxh, out);

    // Phase 2: persistent recurrence over 1024 steps
    rnn_persistent_kernel<<<NWG, 256, 0, stream>>>(Whh, bhh, out, hbuf0, hbuf1, ctr);
}
